// SpatialBottleneck_47244640256452
// MI455X (gfx1250) — matrix operand order repaired and checked by reading, not compiled, not run on hardware
//
#include <hip/hip_runtime.h>
#include <stdint.h>

typedef __attribute__((ext_vector_type(16))) __bf16    v16bf;
typedef __attribute__((ext_vector_type(8)))  float     v8f;
typedef __attribute__((ext_vector_type(4)))  uint32_t  v4u;
typedef __attribute__((ext_vector_type(4)))  uint32_t  v4ui;
typedef __attribute__((ext_vector_type(8)))  int       v8si;
typedef __attribute__((ext_vector_type(4)))  int       v4si;

#if defined(__has_builtin)
#if __has_builtin(__builtin_amdgcn_tensor_load_to_lds) && __has_builtin(__builtin_amdgcn_s_wait_tensorcnt)
#define HAVE_TDM 1
#endif
#endif
#ifndef HAVE_TDM
#define HAVE_TDM 0
#endif

#define N_IMG 32
#define HW    784
#define W_IMG 28
#define CIN   1024
#define CB    256
#define COUT  1024
#define NPIX  (N_IMG * HW)       // 25088 = 196 * 128

#define BM 128
#define BN 128
#define BK 32
#define LDSROW 80                // 64B of bf16 data + 16B pad per LDS row

// workspace layout (bytes)
#define WS_W1B  0u
#define WS_W3B  (512u * 1024u)
#define WS_W2B  (1024u * 1024u)
#define WS_SB1  (WS_W2B + 256u * 2304u * 2u)
#define WS_SB2  (WS_SB1 + 2048u)
#define WS_SB3  (WS_SB2 + 2048u)
#define WS_Y1   (WS_SB3 + 8192u)
#define WS_Y2   (WS_Y1 + (unsigned)NPIX * CB * 2u)

// ---------------------------------------------------------------------------
// Prep: weights -> bf16 (w2 reordered to [co][tap][ci]); BN -> scale/bias
// ---------------------------------------------------------------------------
__global__ void bottleneck_prep(const float* __restrict__ w1,
                                const float* __restrict__ w2,
                                const float* __restrict__ w3,
                                const float* __restrict__ g1, const float* __restrict__ b1,
                                const float* __restrict__ m1, const float* __restrict__ v1,
                                const float* __restrict__ g2, const float* __restrict__ b2,
                                const float* __restrict__ m2, const float* __restrict__ v2,
                                const float* __restrict__ g3, const float* __restrict__ b3,
                                const float* __restrict__ m3, const float* __restrict__ v3,
                                uint8_t* __restrict__ ws)
{
    __bf16* w1b = (__bf16*)(ws + WS_W1B);
    __bf16* w3b = (__bf16*)(ws + WS_W3B);
    __bf16* w2b = (__bf16*)(ws + WS_W2B);
    float*  sb1 = (float*)(ws + WS_SB1);
    float*  sb2 = (float*)(ws + WS_SB2);
    float*  sb3 = (float*)(ws + WS_SB3);

    int id = blockIdx.x * blockDim.x + threadIdx.x;
    if (id < 262144) { w1b[id] = (__bf16)w1[id]; return; }          // [co][ci] already
    id -= 262144;
    if (id < 262144) { w3b[id] = (__bf16)w3[id]; return; }          // [co][ci] already
    id -= 262144;
    if (id < 589824) {                                              // OIHW -> [co][tap][ci]
        int co = id / 2304;
        int r  = id % 2304;
        int t  = r / 256;
        int ci = r % 256;
        w2b[id] = (__bf16)w2[co * 2304 + ci * 9 + t];
        return;
    }
    id -= 589824;
    if (id < 256)  { float s = g1[id] * rsqrtf(v1[id]); sb1[id] = s; sb1[256 + id]  = b1[id] - m1[id] * s; return; }
    id -= 256;
    if (id < 256)  { float s = g2[id] * rsqrtf(v2[id]); sb2[id] = s; sb2[256 + id]  = b2[id] - m2[id] * s; return; }
    id -= 256;
    if (id < 1024) { float s = g3[id] * rsqrtf(v3[id]); sb3[id] = s; sb3[1024 + id] = b3[id] - m3[id] * s; return; }
}

#if HAVE_TDM
// Issue a TDM load of a [128 rows x 32 bf16] tile (row stride = Kelems) into
// LDS at ldsByteAddr, with 16B padding every 64B row (matches LDSROW=80).
// D# layout per CDNA5 ISA; 6-arg builtin (clang-23 form).
__device__ __forceinline__ void tdm_load_w_tile(const __bf16* gsrc,
                                                uint32_t ldsByteAddr, int Kelems)
{
    const uint64_t ga = (uint64_t)(uintptr_t)gsrc;
    v4ui g0;
    g0[0] = 1u;                                               // count=1, user descriptor
    g0[1] = ldsByteAddr;                                      // lds_addr
    g0[2] = (uint32_t)(ga & 0xffffffffu);                     // global_addr[31:0]
    g0[3] = (uint32_t)((ga >> 32) & 0x1ffffffu) | (2u << 30); // global_addr[56:32] | type=2

    const uint32_t dim0 = 32u, dim1 = 128u, t0 = 32u, t1 = 128u;
    const uint32_t s0 = (uint32_t)Kelems;                     // tensor_dim0_stride (elements)
    v8si g1;
    g1[0] = (int)((1u << 16)      // data_size = 2 bytes
                | (1u << 20)      // pad_enable
                | (3u << 22)      // pad_interval: 16 DWORDs (64B)
                | (3u << 25));    // pad_amount:   4 DWORDs (16B)
    g1[1] = (int)((dim0 & 0xffffu) << 16);                    // abar_addr=0 | tensor_dim0 lo
    g1[2] = (int)((dim0 >> 16) | ((dim1 & 0xffffu) << 16));   // tensor_dim0 hi | tensor_dim1 lo
    g1[3] = (int)((dim1 >> 16) | (t0 << 16));                 // tensor_dim1 hi | tile_dim0
    g1[4] = (int)t1;                                          // tile_dim1 | tile_dim2=0
    g1[5] = (int)s0;                                          // tensor_dim0_stride[31:0]
    g1[6] = 0;                                                // stride0 hi | stride1 lo
    g1[7] = 0;
    const v4si gz4 = {0, 0, 0, 0};
    const v8si gz8 = {0, 0, 0, 0, 0, 0, 0, 0};
    __builtin_amdgcn_tensor_load_to_lds(g0, g1, gz4, gz4, gz8, 0);
}
#endif

// ---------------------------------------------------------------------------
// Implicit-GEMM conv via bf16 WMMA.
//   MODE 0: conv1 1x1  (A = x fp32 -> bf16 on the fly, K=1024) -> y1 (bf16)
//   MODE 1: conv2 3x3  (A = y1 shifted per tap, K=2304)        -> y2 (bf16)
//   MODE 2: conv3 1x1  (A = y2, K=256) + identity + relu       -> out (f32)
// Block tile 128x128, BK=32, 8 waves (2x4), wave tile 64x32 (8 WMMA accs).
// Weight tiles are streamed by the Tensor Data Mover (double-buffered LDS).
// ---------------------------------------------------------------------------
template <int MODE>
__global__ __launch_bounds__(256, 2)
void bottleneck_gemm(const float* __restrict__ x,
                     uint8_t* __restrict__ ws,
                     float* __restrict__ out)
{
    constexpr int K    = (MODE == 0) ? CIN : (MODE == 1 ? CB * 9 : CB);
    constexpr int CTOT = (MODE == 2) ? COUT : CB;

    __shared__ __align__(16) uint8_t ldsW[2][BM * LDSROW];
    __shared__ __align__(16) uint8_t ldsA[BN * LDSROW];

    const int tid     = threadIdx.x;
    const int pixBase = blockIdx.x * BN;
    const int coBase  = blockIdx.y * BM;

    const __bf16* Wg  = (const __bf16*)(ws + (MODE == 0 ? WS_W1B : MODE == 1 ? WS_W2B : WS_W3B));
    const float*  sb  = (const float*) (ws + (MODE == 0 ? WS_SB1 : MODE == 1 ? WS_SB2 : WS_SB3));
    const __bf16* Ain = (const __bf16*)(ws + (MODE == 1 ? WS_Y1 : WS_Y2));
    __bf16*       Yo  = (__bf16*)      (ws + (MODE == 0 ? WS_Y1 : WS_Y2));

    // A-tile loader mapping: thread -> (pixel, k-half)
    const int pl    = tid & 127;
    const int khalf = tid >> 7;
    const int pix   = pixBase + pl;
    const int nImg  = pix / HW;
    const int hw    = pix % HW;
    const int hh    = hw / W_IMG;
    const int wwp   = hw % W_IMG;

    // wave / lane mapping (wave32)
    const int lane   = tid & 31;
    const int waveid = tid >> 5;
    const int mW     = (waveid >> 2) * 64;   // wave M offset
    const int nW     = (waveid & 3) * 32;    // wave N offset
    const int lrow   = lane & 15;
    const int lhigh  = lane >> 4;

    v8f acc[4][2] = {};

    const int nK = K / BK;
    const __bf16* Wtile = Wg + (size_t)coBase * K;

#if HAVE_TDM
    const uint32_t ldsWoff0 = (uint32_t)(uintptr_t)&ldsW[0][0];
    const uint32_t ldsWoff1 = (uint32_t)(uintptr_t)&ldsW[1][0];
    if (waveid == 0)
        tdm_load_w_tile(Wtile, ldsWoff0, K);      // prime buffer 0
#endif

    for (int ks = 0; ks < nK; ++ks) {
        const int k0 = ks * BK;

#if !HAVE_TDM
        // fallback: plain global->LDS weight tile (512 x 16B chunks)
#pragma unroll
        for (int i = 0; i < 2; ++i) {
            int idx = tid + i * 256;
            int co  = idx >> 2;
            int kq  = idx & 3;
            v4u val = *(const v4u*)(Wtile + (size_t)co * K + k0 + kq * 8);
            *(v4u*)(ldsW[ks & 1] + co * LDSROW + kq * 16) = val;
        }
#endif

        // ---- activation tile -> LDS, transposed [pix][k] ----
        union { __bf16 h[16]; v4u q[2]; } abuf;
        if constexpr (MODE == 0) {
            const float* src = x + (size_t)nImg * (CIN * HW) + hw;
            const int c0 = k0 + khalf * 16;
#pragma unroll
            for (int j = 0; j < 16; ++j)
                abuf.h[j] = (__bf16)src[(size_t)(c0 + j) * HW];
            if (ks + 1 < nK)
                __builtin_prefetch(src + (size_t)(c0 + BK) * HW, 0, 1);
        } else if constexpr (MODE == 2) {
            const __bf16* src = Ain + (size_t)nImg * (CB * HW) + hw;
            const int c0 = k0 + khalf * 16;
#pragma unroll
            for (int j = 0; j < 16; ++j)
                abuf.h[j] = src[(size_t)(c0 + j) * HW];
            if (ks + 1 < nK)
                __builtin_prefetch(src + (size_t)(c0 + BK) * HW, 0, 1);
        } else {
            const int tap = k0 >> 8;                       // BK=32 divides 256: one tap per step
            const int ci0 = (k0 & 255) + khalf * 16;
            const int hs  = hh + tap / 3 - 1;
            const int wsx = wwp + tap % 3 - 1;
            const bool ok = ((unsigned)hs < 28u) && ((unsigned)wsx < 28u);
            const __bf16* src = Ain + (size_t)nImg * (CB * HW) + hs * W_IMG + wsx;
#pragma unroll
            for (int j = 0; j < 16; ++j)
                abuf.h[j] = ok ? src[(size_t)(ci0 + j) * HW] : (__bf16)0.0f;
        }
        *(v4u*)(ldsA + pl * LDSROW + khalf * 32 + 0)  = abuf.q[0];
        *(v4u*)(ldsA + pl * LDSROW + khalf * 32 + 16) = abuf.q[1];

#if HAVE_TDM
        // Prefetch next weight tile into the other buffer, then make sure the
        // current one has landed before the workgroup barrier releases readers.
        if (waveid == 0) {
            if (ks + 1 < nK) {
                tdm_load_w_tile(Wtile + (size_t)(k0 + BK),
                                ((ks + 1) & 1) ? ldsWoff1 : ldsWoff0, K);
                __builtin_amdgcn_s_wait_tensorcnt(1);   // current tile complete
            } else {
                __builtin_amdgcn_s_wait_tensorcnt(0);
            }
        }
#endif
        __syncthreads();

        // ---- fragments (layouts per ISA 7.12.2) + 8 WMMAs ----
        const uint8_t* wbuf = ldsW[ks & 1];
        union Frag { v16bf bf; v4u q[2]; };
        Frag aF[4], bF[2];
#pragma unroll
        for (int mi = 0; mi < 4; ++mi) {
            const uint8_t* p = wbuf + (mW + mi * 16 + lrow) * LDSROW + lhigh * 16;
            aF[mi].q[0] = *(const v4u*)(p);        // K = kbase + 0..7
            aF[mi].q[1] = *(const v4u*)(p + 32);   // K = kbase + 16..23
        }
#pragma unroll
        for (int ni = 0; ni < 2; ++ni) {
            const uint8_t* p = ldsA + (nW + ni * 16 + lrow) * LDSROW + lhigh * 16;
            bF[ni].q[0] = *(const v4u*)(p);        // K = kbase + 0..7
            bF[ni].q[1] = *(const v4u*)(p + 32);   // K = kbase + 16..23
        }
#pragma unroll
        for (int mi = 0; mi < 4; ++mi)
#pragma unroll
            for (int ni = 0; ni < 2; ++ni)
                acc[mi][ni] = __builtin_amdgcn_wmma_f32_16x16x32_bf16(
                    false, aF[mi].bf, false, bF[ni].bf,
                    (short)0, acc[mi][ni], false, false);

        __syncthreads();
    }

    // ---- epilogue: BN scale/bias (+ residual) + ReLU ----
    const int mrow0 = lhigh * 8;   // C/D layout: lanes16-31 hold M=8..15
#pragma unroll
    for (int ni = 0; ni < 2; ++ni) {
        const int opix = pixBase + nW + ni * 16 + lrow;
        const int on   = opix / HW;
        const int ohw  = opix % HW;
#pragma unroll
        for (int mi = 0; mi < 4; ++mi) {
#pragma unroll
            for (int r = 0; r < 8; ++r) {
                const int co = coBase + mW + mi * 16 + mrow0 + r;
                float v = acc[mi][ni][r] * sb[co] + sb[CTOT + co];
                if constexpr (MODE == 2) {
                    const size_t idx = (size_t)on * (COUT * HW) + (size_t)co * HW + ohw;
                    v += x[idx];
                    out[idx] = v > 0.f ? v : 0.f;
                } else {
                    v = v > 0.f ? v : 0.f;
                    Yo[(size_t)on * (CB * HW) + (size_t)co * HW + ohw] = (__bf16)v;
                }
            }
        }
    }
}

// ---------------------------------------------------------------------------
extern "C" void kernel_launch(void* const* d_in, const int* in_sizes, int n_in,
                              void* d_out, int out_size, void* d_ws, size_t ws_size,
                              hipStream_t stream)
{
    const float* x  = (const float*)d_in[0];
    const float* w1 = (const float*)d_in[1];
    const float* w2 = (const float*)d_in[2];
    const float* w3 = (const float*)d_in[3];
    const float* g1 = (const float*)d_in[4],  *b1 = (const float*)d_in[5];
    const float* m1 = (const float*)d_in[6],  *v1 = (const float*)d_in[7];
    const float* g2 = (const float*)d_in[8],  *b2 = (const float*)d_in[9];
    const float* m2 = (const float*)d_in[10], *v2 = (const float*)d_in[11];
    const float* g3 = (const float*)d_in[12], *b3 = (const float*)d_in[13];
    const float* m3 = (const float*)d_in[14], *v3 = (const float*)d_in[15];
    uint8_t* ws = (uint8_t*)d_ws;

    const int prepTotal = 262144 + 262144 + 589824 + 256 + 256 + 1024;
    bottleneck_prep<<<(prepTotal + 255) / 256, 256, 0, stream>>>(
        w1, w2, w3, g1, b1, m1, v1, g2, b2, m2, v2, g3, b3, m3, v3, ws);

    dim3 grid12(NPIX / BN, CB / BM);     // (196, 2)
    bottleneck_gemm<0><<<grid12, 256, 0, stream>>>(x, ws, nullptr);
    bottleneck_gemm<1><<<grid12, 256, 0, stream>>>(x, ws, nullptr);
    dim3 grid3(NPIX / BN, COUT / BM);    // (196, 8)
    bottleneck_gemm<2><<<grid3, 256, 0, stream>>>(x, ws, (float*)d_out);
}
